// LSSGeometricBEV_79611513799009
// MI455X (gfx1250) — matrix operand order repaired and checked by reading, not compiled, not run on hardware
//
#include <hip/hip_runtime.h>
#include <hip/hip_bf16.h>

// ---------------------------------------------------------------------------
// LSS Geometric BEV on gfx1250:
//  - convs as implicit GEMM on v_wmma_f32_16x16x32_bf16
//  - each wave32 owns 16 pixels x ALL outch (NT accumulators, A reused NT x)
//  - weights staged into LDS once per workgroup by the Tensor Data Mover
//    (tensor_load_to_lds + s_wait_tensorcnt), B fragments read via ds_load
//  - BN folded into per-channel scale/bias, scatter via f32 global atomics
// ---------------------------------------------------------------------------

typedef __attribute__((ext_vector_type(16))) __bf16 v16bf;
typedef __attribute__((ext_vector_type(8)))  __bf16 v8bf;
typedef __attribute__((ext_vector_type(8)))  float  v8f;
typedef unsigned int v4u __attribute__((ext_vector_type(4)));
typedef int          v4i __attribute__((ext_vector_type(4)));
typedef int          v8i __attribute__((ext_vector_type(8)));

#define NB     2
#define NV     6
#define NIMG   12          // NB*NV
#define CIN    64
#define HF     32
#define WF     88
#define PIXF   (HF*WF)     // 2816
#define DD     32
#define BEVC   64
#define BH_    128
#define BW_    128
#define PIXB   (BH_*BW_)   // 16384
#define EXT_   50.0f
#define EPS_   1e-5f

__device__ __forceinline__ __bf16 f2bf(float f) {
  unsigned int u = __float_as_uint(f);
  unsigned int r = (u + 0x7FFFu + ((u >> 16) & 1u)) >> 16;
  unsigned short h = (unsigned short)r;
  return __builtin_bit_cast(__bf16, h);
}

// ---------------------------------------------------------------------------
// TDM: DMA a flat run of bf16 elements from global memory into LDS.
// D# built per CDNA5 ISA ch.8: group0 = {count|flags, lds_addr, global_addr,
// type=2}; group1 = {data_size=2B, tensor_dim0=tile_dim0=elems, 1 row}.
// ---------------------------------------------------------------------------
__device__ __forceinline__ void tdm_load_to_lds(const __bf16* gsrc,
                                                unsigned ldsoff, int elems)
{
  unsigned long long ga = (unsigned long long)(size_t)gsrc;
  v4u g0;
  g0[0] = 1u;                                     // count=1, user descriptor
  g0[1] = ldsoff;                                 // lds_addr (bytes)
  g0[2] = (unsigned)ga;                           // global_addr[31:0]
  g0[3] = (unsigned)((ga >> 32) & 0x1FFFFFFull) | (2u << 30);  // addr[56:32] | type=2
  v8i g1;
  g1[0] = (int)(1u << 16);                        // wg_mask=0, data_size=1 (2 bytes)
  g1[1] = (int)(((unsigned)elems & 0xFFFFu) << 16);       // tensor_dim0[15:0]
  g1[2] = (int)((((unsigned)elems >> 16) & 0xFFFFu) | (1u << 16)); // dim0[31:16], tensor_dim1=1
  g1[3] = (int)(((unsigned)elems & 0xFFFFu) << 16);       // tile_dim0 (16-bit, <=65535)
  g1[4] = 1;                                      // tile_dim1 = 1, tile_dim2 = 0
  g1[5] = elems;                                  // tensor_dim0_stride[31:0]
  g1[6] = 0;                                      // stride hi, tensor_dim1_stride
  g1[7] = 0;
  v4i z = {0, 0, 0, 0};
#if defined(__clang_major__) && (__clang_major__ >= 23)
  v8i z8 = {0, 0, 0, 0, 0, 0, 0, 0};
  __builtin_amdgcn_tensor_load_to_lds(g0, g1, z, z, z8, 0);
#else
  __builtin_amdgcn_tensor_load_to_lds(g0, g1, z, z, 0);
#endif
}

// ---------------------------------------------------------------------------
// Weight prep: pack conv weights to bf16 [tap][outc][inc], fold BN into
// per-outchannel scale/bias:  y = conv*s + ((b - m)*s + beta),  s = g*rsqrt(v+eps)
// ---------------------------------------------------------------------------
__global__ void prep_kernel(
    const float* __restrict__ dh_w1, const float* __restrict__ dh_b1,
    const float* __restrict__ dh_g1, const float* __restrict__ dh_be1,
    const float* __restrict__ dh_m1, const float* __restrict__ dh_v1,
    const float* __restrict__ dh_w2, const float* __restrict__ dh_b2,
    const float* __restrict__ fp_w,  const float* __restrict__ fp_b,
    const float* __restrict__ fp_g,  const float* __restrict__ fp_be,
    const float* __restrict__ fp_m,  const float* __restrict__ fp_v,
    const float* __restrict__ rf_w1, const float* __restrict__ rf_b1,
    const float* __restrict__ rf_g1, const float* __restrict__ rf_be1,
    const float* __restrict__ rf_m1, const float* __restrict__ rf_v1,
    const float* __restrict__ rf_w2, const float* __restrict__ rf_b2,
    const float* __restrict__ rf_g2, const float* __restrict__ rf_be2,
    const float* __restrict__ rf_m2, const float* __restrict__ rf_v2,
    __bf16* __restrict__ wp1, __bf16* __restrict__ wp2, __bf16* __restrict__ wfp,
    __bf16* __restrict__ wr1, __bf16* __restrict__ wr2,
    float* __restrict__ sc1,  float* __restrict__ bi1,  float* __restrict__ b2v,
    float* __restrict__ scfp, float* __restrict__ bifp,
    float* __restrict__ scr1, float* __restrict__ bir1,
    float* __restrict__ scr2, float* __restrict__ bir2)
{
  int tid = blockIdx.x * blockDim.x + threadIdx.x;
  int nth = gridDim.x * blockDim.x;
  // 3x3 weights: OIHW -> [tap][o][i]
  for (int idx = tid; idx < 9 * 64 * 64; idx += nth) {
    int tap = idx >> 12;
    int o   = (idx >> 6) & 63;
    int i   = idx & 63;
    int ky = tap / 3, kx = tap % 3;
    size_t src = (((size_t)o * 64 + i) * 3 + ky) * 3 + kx;
    wp1[idx] = f2bf(dh_w1[src]);
    wr1[idx] = f2bf(rf_w1[src]);
    wr2[idx] = f2bf(rf_w2[src]);
  }
  // 1x1 weights: (O,C,1,1) already [o][i]
  for (int idx = tid; idx < 64 * 64; idx += nth) wfp[idx] = f2bf(fp_w[idx]);
  for (int idx = tid; idx < 32 * 64; idx += nth) wp2[idx] = f2bf(dh_w2[idx]);
  // BN folds
  for (int o = tid; o < 64; o += nth) {
    float s;
    s = dh_g1[o] * rsqrtf(dh_v1[o] + EPS_); sc1[o]  = s; bi1[o]  = dh_be1[o] + (dh_b1[o] - dh_m1[o]) * s;
    s = fp_g[o]  * rsqrtf(fp_v[o]  + EPS_); scfp[o] = s; bifp[o] = fp_be[o]  + (fp_b[o]  - fp_m[o])  * s;
    s = rf_g1[o] * rsqrtf(rf_v1[o] + EPS_); scr1[o] = s; bir1[o] = rf_be1[o] + (rf_b1[o] - rf_m1[o]) * s;
    s = rf_g2[o] * rsqrtf(rf_v2[o] + EPS_); scr2[o] = s; bir2[o] = rf_be2[o] + (rf_b2[o] - rf_m2[o]) * s;
  }
  for (int o = tid; o < 32; o += nth) b2v[o] = dh_b2[o];
}

// NCHW f32 -> [img][pixel][channel] bf16
__global__ void nchw_to_pc_bf16(const float* __restrict__ in, __bf16* __restrict__ out,
                                int nimg, int chans, int pix)
{
  int id = blockIdx.x * blockDim.x + threadIdx.x;
  int total = nimg * chans * pix;
  if (id >= total) return;
  int c = id % chans;
  int p = (id / chans) % pix;
  int n = id / (chans * pix);
  out[id] = f2bf(in[((size_t)n * chans + c) * pix + p]);
}

__global__ void f32_to_bf16(const float* __restrict__ in, __bf16* __restrict__ out, int n)
{
  int id = blockIdx.x * blockDim.x + threadIdx.x;
  if (id < n) out[id] = f2bf(in[id]);
}

__global__ void zero_f32(float* __restrict__ p, int n)
{
  int id = blockIdx.x * blockDim.x + threadIdx.x;
  if (id < n) p[id] = 0.0f;
}

// ---------------------------------------------------------------------------
// Implicit-GEMM conv on WMMA. Activations: bf16 [img][pixel][64].
// Weights: bf16 [tap][OUTC][64], TDM-staged to LDS per workgroup.
// One wave32 per 16-pixel M tile, covering all OUTC (NT accumulators).
// OUTMODE: 0 = bf16 [img][p][OUTC], 1 = f32 [img][p][OUTC], 2 = f32 NCHW.
// Grids must be exact (nimg*P/16 % 4 == 0) so no wave exits before barrier.
// ---------------------------------------------------------------------------
template <int H, int W, int OUTC, bool K3, bool RELU, int OUTMODE>
__global__ __launch_bounds__(128) void conv_wmma_kernel(
    const __bf16* __restrict__ act, const __bf16* __restrict__ wp,
    const float* __restrict__ scalep, const float* __restrict__ biasp,
    void* __restrict__ outp)
{
  constexpr int P    = H * W;
  constexpr int MT   = P / 16;
  constexpr int NT   = OUTC / 16;
  constexpr int NTAP = K3 ? 9 : 1;
  constexpr int WELEMS = NTAP * OUTC * CIN;

  __shared__ __align__(64) __bf16 wlds[WELEMS];

  // Wave 0 DMAs the packed weights into LDS via the Tensor Data Mover.
  if (threadIdx.x < 32) {
    tdm_load_to_lds(wp, (unsigned)(size_t)&wlds[0], WELEMS);
    __builtin_amdgcn_s_wait_tensorcnt(0);
  }
  __syncthreads();

  int wave = blockIdx.x * (blockDim.x >> 5) + (threadIdx.x >> 5);
  int lane = threadIdx.x & 31;
  int img = wave / MT;
  int mt  = wave % MT;
  int pixBase = mt * 16;

  int  ml  = lane & 15;
  bool hi  = lane >= 16;
  int  p0  = pixBase + ml;                    // this lane's A-row (pixel)
  int  px  = p0 % W, py = p0 / W;
  int  kb0 = hi ? 16 : 0;                     // A-fragment half-K base
  int  kbB = hi ? 16 : 0;                     // B-fragment half-K base
  int  wbase = ml * CIN + kbB;                // per-lane B offset in LDS (elements)

  v8f acc[NT];
#pragma unroll
  for (int nt = 0; nt < NT; ++nt) acc[nt] = v8f{0.f, 0.f, 0.f, 0.f, 0.f, 0.f, 0.f, 0.f};

#pragma unroll
  for (int tap = 0; tap < NTAP; ++tap) {
    int dy = K3 ? (tap / 3 - 1) : 0;
    int dx = K3 ? (tap % 3 - 1) : 0;
    int ny = py + dy, nx = px + dx;
    bool valid = (nx >= 0) && (nx < W) && (ny >= 0) && (ny < H);
    const __bf16* arow = act + ((size_t)img * P + (size_t)(ny * W + nx)) * CIN;
#pragma unroll
    for (int ks = 0; ks < 2; ++ks) {          // K = 64 channels = 2 x 32
      v16bf a;
      if (valid) {
        v8bf a0 = *(const v8bf*)(arow + ks * 32 + kb0);        // K = kb0..kb0+7
        v8bf a1 = *(const v8bf*)(arow + ks * 32 + 8 + kb0);    // K = 8+kb0..kb0+15
#pragma unroll
        for (int i = 0; i < 8; ++i) { a[i] = a0[i]; a[8 + i] = a1[i]; }
      } else {
#pragma unroll
        for (int i = 0; i < 16; ++i) a[i] = f2bf(0.0f);
      }
      // reuse A across all N tiles; B comes from LDS (ds_load)
#pragma unroll
      for (int nt = 0; nt < NT; ++nt) {
        v16bf b = *(const v16bf*)&wlds[tap * OUTC * CIN + nt * 16 * CIN + ks * 32 + wbase];
        acc[nt] = __builtin_amdgcn_wmma_f32_16x16x32_bf16(
                      false, a, false, b, (short)0, acc[nt], false, false);
      }
    }
  }

  // epilogue: folded BN + optional ReLU; D layout: VGPR r -> (M = r + hi*8, N = ml)
#pragma unroll
  for (int nt = 0; nt < NT; ++nt) {
    int nc = nt * 16 + ml;
    float s  = scalep ? scalep[nc] : 1.0f;
    float bi = biasp  ? biasp[nc]  : 0.0f;
#pragma unroll
    for (int r = 0; r < 8; ++r) {
      int mr = pixBase + r + (hi ? 8 : 0);
      float v = acc[nt][r] * s + bi;
      if (RELU) v = fmaxf(v, 0.0f);
      if (OUTMODE == 0)
        ((__bf16*)outp)[((size_t)img * P + mr) * OUTC + nc] = f2bf(v);
      else if (OUTMODE == 1)
        ((float*)outp)[((size_t)img * P + mr) * OUTC + nc] = v;
      else
        ((float*)outp)[((size_t)img * OUTC + nc) * P + mr] = v;
    }
  }
}

// Softmax over 32 depth bins, logits/probs layout [img][pixel][32]
__global__ void softmax_kernel(const float* __restrict__ logits, float* __restrict__ probs)
{
  int id = blockIdx.x * blockDim.x + threadIdx.x;
  if (id >= NIMG * PIXF) return;
  const float* L = logits + (size_t)id * DD;
  float m = L[0];
#pragma unroll
  for (int d = 1; d < DD; ++d) m = fmaxf(m, L[d]);
  float e[DD], sum = 0.f;
#pragma unroll
  for (int d = 0; d < DD; ++d) { e[d] = __expf(L[d] - m); sum += e[d]; }
  float inv = 1.0f / sum;
  float* Pr = probs + (size_t)id * DD;
#pragma unroll
  for (int d = 0; d < DD; ++d) Pr[d] = e[d] * inv;
}

// Geometry + scatter: one thread per (view, pixel, depth) point; 64 atomics each.
__global__ void scatter_kernel(
    const float* __restrict__ Kl,   // (B,V,3,3)
    const float* __restrict__ Tl,   // (B,V,4,4)
    const float* __restrict__ trust,// (B,V)
    const float* __restrict__ fp,   // [img][p][64]
    const float* __restrict__ probs,// [img][p][32]
    float* __restrict__ bev)        // [b][cell][64]
{
  int id = blockIdx.x * blockDim.x + threadIdx.x;
  if (id >= NIMG * PIXF * DD) return;
  int d    = id % DD;
  int rest = id / DD;
  int p    = rest % PIXF;
  int n    = rest / PIXF;
  int b    = n / NV;

  const float* K = Kl + n * 9;
  float fx = K[0], cx = K[2], fy = K[4], cy = K[5];
  float x = (float)(p % WF), y = (float)(p / WF);
  // rows of K^-1 pre-scaled by [Wf, Hf, 1]
  float rx = (float)WF * (x - cx) / fx;
  float ry = (float)HF * (y - cy) / fy;
  float depth = 1.0f + 49.0f * (float)d / 31.0f;     // linspace(1, 50, 32)
  float pcx = rx * depth, pcy = ry * depth, pcz = depth;

  const float* T = Tl + n * 16;
  float ex = T[0] * pcx + T[1] * pcy + T[2]  * pcz + T[3];
  float ey = T[4] * pcx + T[5] * pcy + T[6]  * pcz + T[7];

  float s = 2.0f * EXT_;
  int col = (int)((ex + EXT_) / s * (float)(BW_ - 1));   // trunc == astype(int32)
  int row = (int)((ey + EXT_) / s * (float)(BH_ - 1));
  if (col < 0 || col >= BW_ || row < 0 || row >= BH_) return;

  float w = trust[n] * probs[((size_t)n * PIXF + p) * DD + d];
  if (w == 0.0f) return;

  float* dst = bev + ((size_t)b * PIXB + (size_t)(row * BW_ + col)) * BEVC;
  const float* src = fp + ((size_t)n * PIXF + p) * BEVC;
#pragma unroll 8
  for (int c = 0; c < BEVC; ++c) atomicAdd(dst + c, src[c] * w);
}

// ---------------------------------------------------------------------------
extern "C" void kernel_launch(void* const* d_in, const int* in_sizes, int n_in,
                              void* d_out, int out_size, void* d_ws, size_t ws_size,
                              hipStream_t stream)
{
  (void)in_sizes; (void)n_in; (void)out_size; (void)ws_size;
  const float* feat  = (const float*)d_in[0];
  const float* Kl    = (const float*)d_in[1];
  const float* Tl    = (const float*)d_in[2];
  const float* trust = (const float*)d_in[3];
  const float* pr[26];
  for (int i = 0; i < 26; ++i) pr[i] = (const float*)d_in[4 + i];
  // params order: dh_w1 dh_b1 dh_g1 dh_be1 dh_m1 dh_v1 dh_w2 dh_b2
  //               fp_w fp_b fp_g fp_be fp_m fp_v
  //               rf_w1 rf_b1 rf_g1 rf_be1 rf_m1 rf_v1 rf_w2 rf_b2 rf_g2 rf_be2 rf_m2 rf_v2

  // ---- workspace carve-out (256B aligned) ----
  char* ws = (char*)d_ws;
  size_t off = 0;
  auto take = [&](size_t bytes) { void* p = ws + off; off += (bytes + 255) & ~(size_t)255; return p; };
  __bf16* actA  = (__bf16*)take((size_t)NIMG * PIXF * CIN * 2);   // fm, [p][c] bf16
  __bf16* hA    = (__bf16*)take((size_t)NIMG * PIXF * CIN * 2);   // depth-head hidden
  float*  logit = (float*) take((size_t)NIMG * PIXF * DD  * 4);
  float*  probs = (float*) take((size_t)NIMG * PIXF * DD  * 4);
  float*  fpbuf = (float*) take((size_t)NIMG * PIXF * BEVC * 4);
  float*  bev   = (float*) take((size_t)NB * PIXB * BEVC * 4);
  __bf16* bevA  = (__bf16*)take((size_t)NB * PIXB * BEVC * 2);
  __bf16* r1A   = (__bf16*)take((size_t)NB * PIXB * BEVC * 2);
  __bf16* wp1   = (__bf16*)take(9 * 64 * 64 * 2);
  __bf16* wp2   = (__bf16*)take(32 * 64 * 2);
  __bf16* wfp   = (__bf16*)take(64 * 64 * 2);
  __bf16* wr1   = (__bf16*)take(9 * 64 * 64 * 2);
  __bf16* wr2   = (__bf16*)take(9 * 64 * 64 * 2);
  float*  sc1   = (float*)take(64 * 4);  float* bi1  = (float*)take(64 * 4);
  float*  b2v   = (float*)take(32 * 4);
  float*  scfp  = (float*)take(64 * 4);  float* bifp = (float*)take(64 * 4);
  float*  scr1  = (float*)take(64 * 4);  float* bir1 = (float*)take(64 * 4);
  float*  scr2  = (float*)take(64 * 4);  float* bir2 = (float*)take(64 * 4);

  auto cdiv = [](int a, int b) { return (a + b - 1) / b; };

  // 1) weight prep + BN fold
  hipLaunchKernelGGL(prep_kernel, dim3(64), dim3(256), 0, stream,
      pr[0], pr[1], pr[2], pr[3], pr[4], pr[5], pr[6], pr[7],
      pr[8], pr[9], pr[10], pr[11], pr[12], pr[13],
      pr[14], pr[15], pr[16], pr[17], pr[18], pr[19],
      pr[20], pr[21], pr[22], pr[23], pr[24], pr[25],
      wp1, wp2, wfp, wr1, wr2,
      sc1, bi1, b2v, scfp, bifp, scr1, bir1, scr2, bir2);

  // 2) feat_maps NCHW f32 -> [img][p][c] bf16
  {
    int total = NIMG * CIN * PIXF;
    hipLaunchKernelGGL(nchw_to_pc_bf16, dim3(cdiv(total, 256)), dim3(256), 0, stream,
                       feat, actA, NIMG, CIN, PIXF);
  }

  // waves = M tiles only (all outch per wave); all counts divide 4 exactly
  const int wavesF = NIMG * (PIXF / 16);   // 2112
  const int wavesB = NB * (PIXB / 16);     // 2048

  // 3) depth-head conv1 3x3 + BN + ReLU  (WMMA, TDM weights)
  hipLaunchKernelGGL((conv_wmma_kernel<HF, WF, 64, true, true, 0>),
                     dim3(wavesF / 4), dim3(128), 0, stream,
                     actA, wp1, sc1, bi1, (void*)hA);
  // 4) depth logits 1x1 (64 -> 32)  (WMMA, TDM weights)
  hipLaunchKernelGGL((conv_wmma_kernel<HF, WF, 32, false, false, 1>),
                     dim3(wavesF / 4), dim3(128), 0, stream,
                     hA, wp2, (const float*)nullptr, b2v, (void*)logit);
  // 5) softmax over depth bins
  hipLaunchKernelGGL(softmax_kernel, dim3(cdiv(NIMG * PIXF, 256)), dim3(256), 0, stream,
                     logit, probs);
  // 6) feature projection 1x1 + BN + ReLU  (WMMA, TDM weights)
  hipLaunchKernelGGL((conv_wmma_kernel<HF, WF, 64, false, true, 1>),
                     dim3(wavesF / 4), dim3(128), 0, stream,
                     actA, wfp, scfp, bifp, (void*)fpbuf);
  // 7) zero BEV accumulator
  {
    int total = NB * PIXB * BEVC;
    hipLaunchKernelGGL(zero_f32, dim3(cdiv(total, 256)), dim3(256), 0, stream, bev, total);
  }
  // 8) geometry + scatter-add
  {
    int total = NIMG * PIXF * DD;
    hipLaunchKernelGGL(scatter_kernel, dim3(cdiv(total, 256)), dim3(256), 0, stream,
                       Kl, Tl, trust, fpbuf, probs, bev);
  }
  // 9) BEV f32 -> bf16 for refine convs
  {
    int total = NB * PIXB * BEVC;
    hipLaunchKernelGGL(f32_to_bf16, dim3(cdiv(total, 256)), dim3(256), 0, stream,
                       bev, bevA, total);
  }
  // 10) refine conv1 3x3 + BN + ReLU  (WMMA, TDM weights)
  hipLaunchKernelGGL((conv_wmma_kernel<BH_, BW_, 64, true, true, 0>),
                     dim3(wavesB / 4), dim3(128), 0, stream,
                     bevA, wr1, scr1, bir1, (void*)r1A);
  // 11) refine conv2 3x3 + BN + ReLU -> NCHW f32 output  (WMMA, TDM weights)
  hipLaunchKernelGGL((conv_wmma_kernel<BH_, BW_, 64, true, true, 2>),
                     dim3(wavesB / 4), dim3(128), 0, stream,
                     r1A, wr2, scr2, bir2, d_out);
}
